// PyTorchElman_49804440764811
// MI455X (gfx1250) — compile-verified, matrix operand order checked by reading, not run on hardware
//
#include <hip/hip_runtime.h>
#include <stdint.h>

// ---------------- problem constants ----------------
#define S_LEN 2048
#define BATCH 64
#define HID   300
#define HP    320          // hidden padded to 20*16 (=> 10 k-steps of 32)
#define NCLS  2

// ---------------- workspace layout (bytes) ----------------
// W_ih bf16 [320][320]  : 204800
// W_hh bf16 [320][320]  : 204800
// bias  f32 [320]       : 1280
// pooled f32 [2048][320]: 2621440
// xp bf16 [2048][320][64]: 83886080   (n-major, batch contiguous)
#define OFF_WIH  ((size_t)0)
#define OFF_WHH  ((size_t)204800)
#define OFF_BIAS ((size_t)409600)
#define OFF_POOL ((size_t)410880)
#define OFF_XP   ((size_t)3032320)

typedef __attribute__((ext_vector_type(16))) __bf16 v16bf;
typedef __attribute__((ext_vector_type(8)))  float  v8f;
typedef __attribute__((ext_vector_type(4)))  unsigned int u32x4;

union FragBF { u32x4 q[2]; v16bf v; };          // one 16x32 / 32x16 bf16 operand
union PackBF { u32x4 q; unsigned short us[8]; }; // 8 bf16 <-> 16 bytes

__device__ __forceinline__ unsigned short f2bf(float x) {   // RNE f32->bf16
  unsigned u = __float_as_uint(x);
  unsigned r = 0x7FFFu + ((u >> 16) & 1u);
  return (unsigned short)((u + r) >> 16);
}
__device__ __forceinline__ float bf2f(unsigned short h) {
  return __uint_as_float(((unsigned)h) << 16);
}

// ---------------------------------------------------------------------------
// Kernel 0: pack weights to padded bf16, fold the two biases together.
// ---------------------------------------------------------------------------
__global__ void prep_kernel(const float* __restrict__ Wih, const float* __restrict__ Whh,
                            const float* __restrict__ bih, const float* __restrict__ bhh,
                            unsigned short* __restrict__ wih_bf,
                            unsigned short* __restrict__ whh_bf,
                            float* __restrict__ bias) {
  int gid = blockIdx.x * blockDim.x + threadIdx.x;
  int stride = gridDim.x * blockDim.x;
  for (int t = gid; t < HP * HP; t += stride) {
    int r = t / HP, k = t - r * HP;
    bool in = (r < HID) & (k < HID);
    wih_bf[t] = in ? f2bf(Wih[r * HID + k]) : (unsigned short)0;
    whh_bf[t] = in ? f2bf(Whh[r * HID + k]) : (unsigned short)0;
  }
  if (gid < HP) bias[gid] = (gid < HID) ? (bih[gid] + bhh[gid]) : 0.f;
}

// ---------------------------------------------------------------------------
// Kernel 1: xp[s] = emb[x[s]] @ W_ih^T + (b_ih+b_hh), one WG per timestep.
// 512 threads = 16 waves; each wave owns 5 of the 4x20 16x16 output tiles.
// A (gathered embeddings, bf16) lives in LDS; B (W_ih rows) streams from L2.
// Output stored transposed [s][n][b] bf16 so each lane stores 16B contiguous.
// ---------------------------------------------------------------------------
__global__ void __launch_bounds__(512)
xp_kernel(const int* __restrict__ x, const float* __restrict__ emb,
          const unsigned short* __restrict__ wih_bf, const float* __restrict__ bias,
          unsigned short* __restrict__ xp) {
  extern __shared__ unsigned char smem[];
  unsigned short* As = (unsigned short*)smem;            // [64][320] bf16
  int* tok = (int*)(smem + BATCH * HP * 2);              // [64]
  const int tid = threadIdx.x;
  const int s = blockIdx.x;

  if (tid < BATCH) tok[tid] = x[s * BATCH + tid];
  __syncthreads();
  for (int t = tid; t < BATCH * HP; t += 512) {
    int r = t / HP, k = t - r * HP;
    As[t] = (k < HID) ? f2bf(emb[(size_t)tok[r] * HID + k]) : (unsigned short)0;
  }
  __syncthreads();

  const int w = tid >> 5, l = tid & 31, idx = l & 15, hf = l >> 4;
  const int m0 = (w & 3) * 16, ng = w >> 2;

  for (int u = 0; u < 5; ++u) {
    const int n = (ng + 4 * u) * 16 + idx;               // output hidden index
    float bn = bias[n];
    v8f c;
#pragma unroll
    for (int r = 0; r < 8; ++r) c[r] = bn;
#pragma unroll
    for (int kb = 0; kb < HP; kb += 32) {
      FragBF a, b;
      a.q[0] = *(const u32x4*)(As + (m0 + idx) * HP + kb + hf * 8);
      a.q[1] = *(const u32x4*)(As + (m0 + idx) * HP + kb + 16 + hf * 8);
      b.q[0] = *(const u32x4*)(wih_bf + (size_t)n * HP + kb + hf * 8);
      b.q[1] = *(const u32x4*)(wih_bf + (size_t)n * HP + kb + 16 + hf * 8);
      c = __builtin_amdgcn_wmma_f32_16x16x32_bf16(false, a.v, false, b.v,
                                                  (short)0, c, false, false);
    }
    PackBF p;
#pragma unroll
    for (int r = 0; r < 8; ++r) p.us[r] = f2bf(c[r]);
    // lane holds rows b = m0 + hf*8 .. +7 at column n -> contiguous in [s][n][b]
    *(u32x4*)(xp + ((size_t)s * HP + n) * BATCH + m0 + hf * 8) = p.q;
  }
}

// ---------------------------------------------------------------------------
// Kernel 2: the serial Elman scan. ONE persistent workgroup (1 WGP):
// W_hh bf16 (204.8KB) + h ping-pong bf16 (82KB) + pooled (1.3KB) all in the
// 320KB LDS. Per step: 5 tiles/wave * 10 WMMAs off LDS, xp streamed 16B/lane
// (prefetched one step ahead), relu, ds_max_u32 batch-max, 2 barriers.
// ---------------------------------------------------------------------------
__global__ void __launch_bounds__(512)
rnn_kernel(const unsigned short* __restrict__ whh_bf, const unsigned short* __restrict__ xp,
           float* __restrict__ pooled_g, float* __restrict__ hT_out) {
  extern __shared__ unsigned char smem[];
  unsigned short* W   = (unsigned short*)smem;                         // [320][320]
  unsigned short* hb0 = (unsigned short*)(smem + (size_t)HP * HP * 2); // [64][320]
  unsigned short* hb1 = hb0 + BATCH * HP;
  unsigned int*   pool = (unsigned int*)(smem + (size_t)HP * HP * 2 + 2u * BATCH * HP * 2);
  const int tid = threadIdx.x;

  { // resident init: W_hh -> LDS, zero h buffers, zero pooled
    const u32x4* src = (const u32x4*)whh_bf;
    u32x4* dst = (u32x4*)W;
    for (int i = tid; i < HP * HP * 2 / 16; i += 512) dst[i] = src[i];
    u32x4 z = {0u, 0u, 0u, 0u};
    u32x4* hz = (u32x4*)hb0;
    for (int i = tid; i < 2 * BATCH * HP * 2 / 16; i += 512) hz[i] = z;
    if (tid < HP) pool[tid] = 0u;
  }
  __syncthreads();

  const int w = tid >> 5, l = tid & 31, idx = l & 15, hf = l >> 4;
  const int m0 = (w & 3) * 16, ng = w >> 2;
  const int b0 = m0 + hf * 8;

  for (int s = 0; s < S_LEN; ++s) {
    const unsigned short* hc = (s & 1) ? hb1 : hb0;
    unsigned short*       hn = (s & 1) ? hb0 : hb1;

    for (int u = 0; u < 5; ++u) {
      const int n = (ng + 4 * u) * 16 + idx;
      const size_t xoff = ((size_t)s * HP + n) * BATCH + b0;
      if (s + 1 < S_LEN)                                  // hide next step's xp
        __builtin_prefetch((const void*)(xp + xoff + (size_t)HP * BATCH), 0, 0);

      PackBF xq; xq.q = *(const u32x4*)(xp + xoff);       // C init from xp
      v8f c;
#pragma unroll
      for (int r = 0; r < 8; ++r) c[r] = bf2f(xq.us[r]);

#pragma unroll
      for (int kb = 0; kb < HP; kb += 32) {
        FragBF a, b;
        a.q[0] = *(const u32x4*)(hc + (m0 + idx) * HP + kb + hf * 8);
        a.q[1] = *(const u32x4*)(hc + (m0 + idx) * HP + kb + 16 + hf * 8);
        b.q[0] = *(const u32x4*)(W + n * HP + kb + hf * 8);
        b.q[1] = *(const u32x4*)(W + n * HP + kb + 16 + hf * 8);
        c = __builtin_amdgcn_wmma_f32_16x16x32_bf16(false, a.v, false, b.v,
                                                    (short)0, c, false, false);
      }

      float lmax = 0.f;
#pragma unroll
      for (int r = 0; r < 8; ++r) {
        c[r] = fmaxf(c[r], 0.f);                          // relu
        lmax = fmaxf(lmax, c[r]);
        hn[(b0 + r) * HP + n] = f2bf(c[r]);               // h_{s} for next step
      }
      // post-relu values are >=0 -> IEEE bit pattern is monotone: u32 max works
      atomicMax(&pool[n], __float_as_uint(lmax));
    }
    __syncthreads();
    if (tid < HP) {                                       // drain + reset pooled
      pooled_g[(size_t)s * HP + tid] = __uint_as_float(pool[tid]);
      pool[tid] = 0u;
    }
    __syncthreads();
  }

  // S even -> final hidden state sits in hb0; emit hT (B,H) as f32
  for (int i = tid; i < BATCH * HID; i += 512) {
    int b = i / HID, j = i - b * HID;
    hT_out[i] = bf2f(hb0[b * HP + j]);
  }
}

// ---------------------------------------------------------------------------
// Kernel 3: out[s,c] = pooled[s,:300] . W_out[c,:] + b_out[c]   (4096 dots)
// ---------------------------------------------------------------------------
__global__ void out_kernel(const float* __restrict__ pooled_g,
                           const float* __restrict__ Wout, const float* __restrict__ bout,
                           float* __restrict__ out) {
  int i = blockIdx.x * blockDim.x + threadIdx.x;
  if (i >= S_LEN * NCLS) return;
  int s = i >> 1, c = i & 1;
  const float* p = pooled_g + (size_t)s * HP;
  const float* wv = Wout + c * HID;
  float acc = bout[c];
  for (int k = 0; k < HID; ++k) acc = fmaf(p[k], wv[k], acc);
  out[i] = acc;
}

// ---------------------------------------------------------------------------
extern "C" void kernel_launch(void* const* d_in, const int* in_sizes, int n_in,
                              void* d_out, int out_size, void* d_ws, size_t ws_size,
                              hipStream_t stream) {
  const int*   x    = (const int*)  d_in[0];   // (S,B) int32
  const float* emb  = (const float*)d_in[1];   // (V,H)
  const float* Wih  = (const float*)d_in[2];   // (H,H)
  const float* Whh  = (const float*)d_in[3];   // (H,H)
  const float* bih  = (const float*)d_in[4];   // (H)
  const float* bhh  = (const float*)d_in[5];   // (H)
  const float* Wout = (const float*)d_in[6];   // (C,H)
  const float* bout = (const float*)d_in[7];   // (C)

  float* out = (float*)d_out;                  // (S,C) = 4096 floats ...
  float* hT  = out + S_LEN * NCLS;             // ... then (1,B,H) = 19200 floats

  unsigned char* ws = (unsigned char*)d_ws;    // needs ~87 MB (xp dominates)
  unsigned short* wih_bf  = (unsigned short*)(ws + OFF_WIH);
  unsigned short* whh_bf  = (unsigned short*)(ws + OFF_WHH);
  float*          bias    = (float*)         (ws + OFF_BIAS);
  float*          pooled  = (float*)         (ws + OFF_POOL);
  unsigned short* xp      = (unsigned short*)(ws + OFF_XP);

  // 0) pack weights to padded bf16
  prep_kernel<<<dim3(256), dim3(256), 0, stream>>>(Wih, Whh, bih, bhh,
                                                   wih_bf, whh_bf, bias);

  // 1) xp for all timesteps (parallel GEMM, WMMA bf16)
  size_t xp_lds = (size_t)BATCH * HP * 2 + BATCH * sizeof(int);   // 41216 B
  xp_kernel<<<dim3(S_LEN), dim3(512), xp_lds, stream>>>(x, emb, wih_bf, bias, xp);

  // 2) serial scan: one persistent WGP, everything LDS-resident (288000 B)
  size_t rnn_lds = (size_t)HP * HP * 2 + 2u * BATCH * HP * 2 + HP * 4;
  rnn_kernel<<<dim3(1), dim3(512), rnn_lds, stream>>>(whh_bf, xp, pooled, hT);

  // 3) tiny output head
  out_kernel<<<dim3((S_LEN * NCLS + 255) / 256), dim3(256), 0, stream>>>(
      pooled, Wout, bout, out);
}
